// SelfAttention_57784490000759
// MI455X (gfx1250) — compile-verified, matrix operand order checked by reading, not run on hardware
//
#include <hip/hip_runtime.h>

#define BATCH   8
#define CCH     512
#define NPIX    1024
#define HEADS_N 8
#define DHEAD   64
#define NGROUP  32
#define CPG     16          // channels per group
#define EPS_    1e-5f
#define SCALE_  0.125f      // d^-0.5, d=64

typedef __bf16 bf16_t;
typedef __attribute__((ext_vector_type(16))) __bf16 v16bf;
typedef __attribute__((ext_vector_type(8)))  __bf16 v8bf;
typedef __attribute__((ext_vector_type(8)))  float  v8f;
typedef int v4i_t __attribute__((vector_size(16)));

#if defined(__AMDGCN__) && __has_builtin(__builtin_amdgcn_global_load_async_to_lds_b128) && \
    __has_builtin(__builtin_amdgcn_s_wait_asynccnt)
#define HAS_ASYNC_LDS 1
typedef __attribute__((address_space(1))) v4i_t* as1_v4i;
typedef __attribute__((address_space(3))) v4i_t* as3_v4i;
#else
#define HAS_ASYNC_LDS 0
#endif

union Frag16 { v16bf v; v8bf h[2]; };

__device__ __forceinline__ v16bf load_frag(const bf16_t* base, int kb) {
  // per-lane A/B fragment: 8 bf16 at +kb, 8 bf16 at +kb+16 (kb in {0,8})
  Frag16 f;
  f.h[0] = *(const v8bf*)(base + kb);
  f.h[1] = *(const v8bf*)(base + kb + 16);
  return f.v;
}

// one 16-byte global -> LDS chunk; async on CDNA5, register-staged fallback otherwise
__device__ __forceinline__ void lds_cp16(bf16_t* lds_dst, const bf16_t* gsrc) {
#if HAS_ASYNC_LDS
  __builtin_amdgcn_global_load_async_to_lds_b128((as1_v4i)gsrc, (as3_v4i)lds_dst, 0, 0);
#else
  *(uint4*)lds_dst = *(const uint4*)gsrc;
#endif
}

__device__ __forceinline__ void lds_cp_join() {
#if HAS_ASYNC_LDS
  __builtin_amdgcn_s_wait_asynccnt(0);
#endif
}

__device__ __forceinline__ v8f wmma_bf16(v16bf a, v16bf b, v8f c) {
  return __builtin_amdgcn_wmma_f32_16x16x32_bf16(false, a, false, b, (short)0, c,
                                                 false, false);
}

// ---------------------------------------------------------------- weights f32->bf16
__global__ __launch_bounds__(256) void convert_w_kernel(const float* __restrict__ wq,
                                                        const float* __restrict__ wp,
                                                        bf16_t* __restrict__ wq_b,
                                                        bf16_t* __restrict__ wp_b) {
  const int nq = 3 * CCH * CCH;
  const int np = CCH * CCH;
  int idx = blockIdx.x * 256 + threadIdx.x;
  if (idx < nq) {
    wq_b[idx] = (bf16_t)wq[idx];
  } else if (idx < nq + np) {
    int j = idx - nq;
    wp_b[j] = (bf16_t)wp[j];
  }
}

// ---------------------------------------------------------------- GroupNorm -> xn[B][N][C] bf16
__global__ __launch_bounds__(256) void groupnorm_kernel(const float* __restrict__ x,
                                                        const float* __restrict__ gw,
                                                        const float* __restrict__ gb,
                                                        bf16_t* __restrict__ xn) {
  const int b = blockIdx.x / NGROUP;
  const int g = blockIdx.x % NGROUP;
  const int tid = threadIdx.x;
  const float* xp = x + (size_t)b * CCH * NPIX + (size_t)g * CPG * NPIX;
  const int nelem = CPG * NPIX;  // 16384

  float s = 0.0f, ss = 0.0f;
  for (int i = tid; i < nelem; i += 256) {
    float v = xp[i];
    s  += v;
    ss += v * v;
  }
  __shared__ float rs[256], rss[256];
  rs[tid]  = s;
  rss[tid] = ss;
  __syncthreads();
  for (int off = 128; off > 0; off >>= 1) {
    if (tid < off) { rs[tid] += rs[tid + off]; rss[tid] += rss[tid + off]; }
    __syncthreads();
  }
  const float inv_n = 1.0f / (float)nelem;
  const float mean  = rs[0] * inv_n;
  const float var   = rss[0] * inv_n - mean * mean;
  const float rstd  = rsqrtf(var + EPS_);

  for (int i = tid; i < nelem; i += 256) {
    int cc = i >> 10;          // i / NPIX
    int n  = i & (NPIX - 1);
    int c  = g * CPG + cc;
    float v = (xp[i] - mean) * rstd * gw[c] + gb[c];
    xn[((size_t)b * NPIX + n) * CCH + c] = (bf16_t)v;
  }
}

// ---------------------------------------------------------------- QKV GEMM (per batch): W[3C x C] * Xn^T
// A = w_qkv bf16 [1536][512] row-major. B = xn [N][C] (K-contiguous per column n).
// Scatter: q,k -> [B][H][N][D]; v -> transposed [B][H][D][N].
__global__ __launch_bounds__(256) void qkv_gemm_kernel(const bf16_t* __restrict__ Wq,
                                                       const bf16_t* __restrict__ Xn,
                                                       bf16_t* __restrict__ Qo,
                                                       bf16_t* __restrict__ Ko,
                                                       bf16_t* __restrict__ Vt) {
  __shared__ __attribute__((aligned(16))) bf16_t As[2][128 * 48];
  __shared__ __attribute__((aligned(16))) bf16_t Bs[2][128 * 48];
  const int b    = blockIdx.z;
  const int m0   = blockIdx.y * 128;
  const int n0   = blockIdx.x * 128;
  const int tid  = threadIdx.x;
  const int lane = tid & 31;
  const int wave = tid >> 5;
  const int wy   = wave & 3;   // M subtiles (32 rows)
  const int wx   = wave >> 2;  // N subtiles (64 cols)
  const int r16  = lane & 15;
  const int kb   = (lane >> 4) * 8;
  const int mhalf = (lane >> 4) * 8;
  const bf16_t* Bbase = Xn + (size_t)b * NPIX * CCH;

  // staging chunk assignment: thread tid covers rows r0 and r0+64, kc0 fixed
  const int r0  = tid >> 2;
  const int kc0 = (tid & 3) * 8;

  auto stage = [&](int buf, int k0) {
    lds_cp16(&As[buf][r0 * 48 + kc0],        Wq + (size_t)(m0 + r0) * CCH + k0 + kc0);
    lds_cp16(&As[buf][(r0 + 64) * 48 + kc0], Wq + (size_t)(m0 + r0 + 64) * CCH + k0 + kc0);
    lds_cp16(&Bs[buf][r0 * 48 + kc0],        Bbase + (size_t)(n0 + r0) * CCH + k0 + kc0);
    lds_cp16(&Bs[buf][(r0 + 64) * 48 + kc0], Bbase + (size_t)(n0 + r0 + 64) * CCH + k0 + kc0);
  };

  const v8f vzero = {0.f, 0.f, 0.f, 0.f, 0.f, 0.f, 0.f, 0.f};
  v8f acc[2][4];
  for (int i = 0; i < 2; ++i)
    for (int j = 0; j < 4; ++j) acc[i][j] = vzero;

  stage(0, 0);
  lds_cp_join();
  __syncthreads();

  int cur = 0;
#pragma unroll 2
  for (int k0 = 0; k0 < CCH; k0 += 32) {
    if (k0 + 32 < CCH) {
      stage(cur ^ 1, k0 + 32);  // async: overlaps with WMMAs below
#if !HAS_ASYNC_LDS
      __builtin_prefetch(Wq + (size_t)(m0 + r0) * CCH + k0 + 64, 0, 1);
#endif
    }
    v16bf fa[2], fb[4];
    for (int mt = 0; mt < 2; ++mt)
      fa[mt] = load_frag(&As[cur][(wy * 32 + mt * 16 + r16) * 48], kb);
    for (int nt = 0; nt < 4; ++nt)
      fb[nt] = load_frag(&Bs[cur][(wx * 64 + nt * 16 + r16) * 48], kb);
    for (int mt = 0; mt < 2; ++mt)
      for (int nt = 0; nt < 4; ++nt)
        acc[mt][nt] = wmma_bf16(fa[mt], fb[nt], acc[mt][nt]);
    lds_cp_join();
    __syncthreads();
    cur ^= 1;
  }

  for (int mt = 0; mt < 2; ++mt) {
    for (int nt = 0; nt < 4; ++nt) {
      const int n = n0 + wx * 64 + nt * 16 + r16;
#pragma unroll
      for (int r = 0; r < 8; ++r) {
        int o = m0 + wy * 32 + mt * 16 + mhalf + r;
        bf16_t val = (bf16_t)acc[mt][nt][r];
        int which = o >> 9;
        int oo = o & 511;
        int h = oo >> 6;
        int dd = oo & 63;
        if (which == 0)
          Qo[(((size_t)b * HEADS_N + h) * NPIX + n) * DHEAD + dd] = val;
        else if (which == 1)
          Ko[(((size_t)b * HEADS_N + h) * NPIX + n) * DHEAD + dd] = val;
        else
          Vt[(((size_t)b * HEADS_N + h) * DHEAD + dd) * NPIX + n] = val;
      }
    }
  }
}

// ---------------------------------------------------------------- Flash attention per (b, h, 64-query tile)
__global__ __launch_bounds__(256) void attn_kernel(const bf16_t* __restrict__ Q,
                                                   const bf16_t* __restrict__ K,
                                                   const bf16_t* __restrict__ Vt,
                                                   bf16_t* __restrict__ Attn) {
  __shared__ __attribute__((aligned(16))) bf16_t Qs[64 * 72];
  __shared__ __attribute__((aligned(16))) bf16_t Ks[64 * 72];
  __shared__ __attribute__((aligned(16))) bf16_t Vs[64 * 72];
  __shared__ __attribute__((aligned(16))) bf16_t Ps[64 * 72];
  __shared__ float Ss[64 * 66];
  __shared__ float mrow[64], lrow[64], srow[64];

  const int b  = blockIdx.z;
  const int h  = blockIdx.y;
  const int q0 = blockIdx.x * 64;
  const int tid  = threadIdx.x;
  const int lane = tid & 31;
  const int wave = tid >> 5;
  const int wy   = wave & 3;   // 16 query rows
  const int wx   = wave >> 2;  // 32 key cols / 32 d cols
  const int r16  = lane & 15;
  const int kb   = (lane >> 4) * 8;
  const int mhalf = (lane >> 4) * 8;

  const bf16_t* Qg = Q + (size_t)(b * HEADS_N + h) * NPIX * DHEAD;
  const bf16_t* Kg = K + (size_t)(b * HEADS_N + h) * NPIX * DHEAD;
  const bf16_t* Vg = Vt + (size_t)(b * HEADS_N + h) * DHEAD * NPIX;

  // staging: thread covers rows tr0 and tr0+32 of a 64x64 bf16 tile (8 chunks/row)
  const int tr0 = tid >> 3;
  const int tkc = (tid & 7) * 8;

  lds_cp16(&Qs[tr0 * 72 + tkc],        Qg + (size_t)(q0 + tr0) * DHEAD + tkc);
  lds_cp16(&Qs[(tr0 + 32) * 72 + tkc], Qg + (size_t)(q0 + tr0 + 32) * DHEAD + tkc);
  if (tid < 64) { mrow[tid] = -1e30f; lrow[tid] = 0.0f; }
  lds_cp_join();
  __syncthreads();

  v16bf fq0 = load_frag(&Qs[(wy * 16 + r16) * 72], kb);
  v16bf fq1 = load_frag(&Qs[(wy * 16 + r16) * 72 + 32], kb);

  const v8f vzero = {0.f, 0.f, 0.f, 0.f, 0.f, 0.f, 0.f, 0.f};
  v8f oacc[2];
  oacc[0] = vzero;
  oacc[1] = vzero;

  for (int j0 = 0; j0 < NPIX; j0 += 64) {
    __syncthreads();  // previous P*V reads of Ks/Vs/Ps done before restaging
    lds_cp16(&Ks[tr0 * 72 + tkc],        Kg + (size_t)(j0 + tr0) * DHEAD + tkc);        // [j][d]
    lds_cp16(&Ks[(tr0 + 32) * 72 + tkc], Kg + (size_t)(j0 + tr0 + 32) * DHEAD + tkc);
    lds_cp16(&Vs[tr0 * 72 + tkc],        Vg + (size_t)tr0 * NPIX + j0 + tkc);           // [d][j]
    lds_cp16(&Vs[(tr0 + 32) * 72 + tkc], Vg + (size_t)(tr0 + 32) * NPIX + j0 + tkc);
    lds_cp_join();
    __syncthreads();

    // S tile = Q * K^T  (raw scores -> LDS f32)
    for (int nt = 0; nt < 2; ++nt) {
      const int jloc = wx * 32 + nt * 16 + r16;
      v8f s = vzero;
      s = wmma_bf16(fq0, load_frag(&Ks[jloc * 72], kb), s);
      s = wmma_bf16(fq1, load_frag(&Ks[jloc * 72 + 32], kb), s);
#pragma unroll
      for (int r = 0; r < 8; ++r)
        Ss[(wy * 16 + mhalf + r) * 66 + wx * 32 + nt * 16 + r16] = s[r];
    }
    __syncthreads();

    // online softmax, one thread per query row
    if (tid < 64) {
      const float* sp = &Ss[tid * 66];
      float m_old = mrow[tid];
      float mx = m_old;
#pragma unroll 4
      for (int j = 0; j < 64; ++j) {
        float v = sp[j] * SCALE_;
        mx = v > mx ? v : mx;
      }
      float fac = __expf(m_old - mx);
      float lsum = 0.0f;
#pragma unroll 4
      for (int j = 0; j < 64; ++j) {
        float p = __expf(sp[j] * SCALE_ - mx);
        lsum += p;
        Ps[tid * 72 + j] = (bf16_t)p;
      }
      mrow[tid] = mx;
      lrow[tid] = lrow[tid] * fac + lsum;
      srow[tid] = fac;
    }
    __syncthreads();

    // rescale O, then O += P * V
#pragma unroll
    for (int nt = 0; nt < 2; ++nt)
#pragma unroll
      for (int r = 0; r < 8; ++r)
        oacc[nt][r] *= srow[wy * 16 + mhalf + r];

    v16bf fp0 = load_frag(&Ps[(wy * 16 + r16) * 72], kb);
    v16bf fp1 = load_frag(&Ps[(wy * 16 + r16) * 72 + 32], kb);
    for (int nt = 0; nt < 2; ++nt) {
      const int ddloc = wx * 32 + nt * 16 + r16;
      oacc[nt] = wmma_bf16(fp0, load_frag(&Vs[ddloc * 72], kb), oacc[nt]);
      oacc[nt] = wmma_bf16(fp1, load_frag(&Vs[ddloc * 72 + 32], kb), oacc[nt]);
    }
  }

  // normalize and write attn output [B][N][C], c = h*64 + dd
  for (int nt = 0; nt < 2; ++nt) {
    const int dd = wx * 32 + nt * 16 + r16;
#pragma unroll
    for (int r = 0; r < 8; ++r) {
      int m = wy * 16 + mhalf + r;
      float val = oacc[nt][r] / lrow[m];
      Attn[((size_t)b * NPIX + q0 + m) * CCH + h * DHEAD + dd] = (bf16_t)val;
    }
  }
}

// ---------------------------------------------------------------- Proj GEMM + residual
__global__ __launch_bounds__(256) void proj_gemm_kernel(const bf16_t* __restrict__ Wp,
                                                        const bf16_t* __restrict__ Xa,
                                                        const float* __restrict__ Xres,
                                                        float* __restrict__ Out) {
  __shared__ __attribute__((aligned(16))) bf16_t As[2][128 * 48];
  __shared__ __attribute__((aligned(16))) bf16_t Bs[2][128 * 48];
  const int b    = blockIdx.z;
  const int m0   = blockIdx.y * 128;
  const int n0   = blockIdx.x * 128;
  const int tid  = threadIdx.x;
  const int lane = tid & 31;
  const int wave = tid >> 5;
  const int wy   = wave & 3;
  const int wx   = wave >> 2;
  const int r16  = lane & 15;
  const int kb   = (lane >> 4) * 8;
  const int mhalf = (lane >> 4) * 8;
  const bf16_t* Bbase = Xa + (size_t)b * NPIX * CCH;

  const int r0  = tid >> 2;
  const int kc0 = (tid & 3) * 8;

  auto stage = [&](int buf, int k0) {
    lds_cp16(&As[buf][r0 * 48 + kc0],        Wp + (size_t)(m0 + r0) * CCH + k0 + kc0);
    lds_cp16(&As[buf][(r0 + 64) * 48 + kc0], Wp + (size_t)(m0 + r0 + 64) * CCH + k0 + kc0);
    lds_cp16(&Bs[buf][r0 * 48 + kc0],        Bbase + (size_t)(n0 + r0) * CCH + k0 + kc0);
    lds_cp16(&Bs[buf][(r0 + 64) * 48 + kc0], Bbase + (size_t)(n0 + r0 + 64) * CCH + k0 + kc0);
  };

  const v8f vzero = {0.f, 0.f, 0.f, 0.f, 0.f, 0.f, 0.f, 0.f};
  v8f acc[2][4];
  for (int i = 0; i < 2; ++i)
    for (int j = 0; j < 4; ++j) acc[i][j] = vzero;

  stage(0, 0);
  lds_cp_join();
  __syncthreads();

  int cur = 0;
#pragma unroll 2
  for (int k0 = 0; k0 < CCH; k0 += 32) {
    if (k0 + 32 < CCH) {
      stage(cur ^ 1, k0 + 32);
#if !HAS_ASYNC_LDS
      __builtin_prefetch(Wp + (size_t)(m0 + r0) * CCH + k0 + 64, 0, 1);
#endif
    }
    v16bf fa[2], fb[4];
    for (int mt = 0; mt < 2; ++mt)
      fa[mt] = load_frag(&As[cur][(wy * 32 + mt * 16 + r16) * 48], kb);
    for (int nt = 0; nt < 4; ++nt)
      fb[nt] = load_frag(&Bs[cur][(wx * 64 + nt * 16 + r16) * 48], kb);
    for (int mt = 0; mt < 2; ++mt)
      for (int nt = 0; nt < 4; ++nt)
        acc[mt][nt] = wmma_bf16(fa[mt], fb[nt], acc[mt][nt]);
    lds_cp_join();
    __syncthreads();
    cur ^= 1;
  }

  for (int mt = 0; mt < 2; ++mt) {
    for (int nt = 0; nt < 4; ++nt) {
      const int n = n0 + wx * 64 + nt * 16 + r16;
#pragma unroll
      for (int r = 0; r < 8; ++r) {
        int o = m0 + wy * 32 + mt * 16 + mhalf + r;
        size_t idx = ((size_t)b * CCH + o) * NPIX + n;
        Out[idx] = Xres[idx] + acc[mt][nt][r];
      }
    }
  }
}

// ---------------------------------------------------------------- launch
extern "C" void kernel_launch(void* const* d_in, const int* in_sizes, int n_in,
                              void* d_out, int out_size, void* d_ws, size_t ws_size,
                              hipStream_t stream) {
  (void)in_sizes; (void)n_in; (void)out_size; (void)ws_size;
  const float* x      = (const float*)d_in[0];
  const float* gn_w   = (const float*)d_in[1];
  const float* gn_b   = (const float*)d_in[2];
  const float* w_qkv  = (const float*)d_in[3];
  const float* w_proj = (const float*)d_in[4];
  float* out = (float*)d_out;

  char* ws = (char*)d_ws;
  size_t off = 0;
  auto take = [&](size_t bytes) -> void* {
    void* p = ws + off;
    off += (bytes + 255) & ~(size_t)255;
    return p;
  };
  bf16_t* wq_b  = (bf16_t*)take((size_t)3 * CCH * CCH * sizeof(bf16_t));
  bf16_t* wp_b  = (bf16_t*)take((size_t)CCH * CCH * sizeof(bf16_t));
  bf16_t* xn    = (bf16_t*)take((size_t)BATCH * NPIX * CCH * sizeof(bf16_t));
  bf16_t* q_b   = (bf16_t*)take((size_t)BATCH * HEADS_N * NPIX * DHEAD * sizeof(bf16_t));
  bf16_t* k_b   = (bf16_t*)take((size_t)BATCH * HEADS_N * NPIX * DHEAD * sizeof(bf16_t));
  bf16_t* vt_b  = (bf16_t*)take((size_t)BATCH * HEADS_N * DHEAD * NPIX * sizeof(bf16_t));
  bf16_t* at_b  = (bf16_t*)take((size_t)BATCH * NPIX * CCH * sizeof(bf16_t));

  const int nconv = 3 * CCH * CCH + CCH * CCH;
  convert_w_kernel<<<(nconv + 255) / 256, 256, 0, stream>>>(w_qkv, w_proj, wq_b, wp_b);

  groupnorm_kernel<<<BATCH * NGROUP, 256, 0, stream>>>(x, gn_w, gn_b, xn);

  qkv_gemm_kernel<<<dim3(NPIX / 128, (3 * CCH) / 128, BATCH), 256, 0, stream>>>(
      wq_b, xn, q_b, k_b, vt_b);

  attn_kernel<<<dim3(NPIX / 64, HEADS_N, BATCH), 256, 0, stream>>>(q_b, k_b, vt_b, at_b);

  proj_gemm_kernel<<<dim3(NPIX / 128, CCH / 128, BATCH), 256, 0, stream>>>(
      wp_b, at_b, x, out);
}
